// QuantizationModule_6399501271795
// MI455X (gfx1250) — compile-verified, matrix operand order checked by reading, not run on hardware
//
#include <hip/hip_runtime.h>

// ---------------------------------------------------------------------------
// Gumbel-softmax VQ forward for MI455X (gfx1250, wave32, WMMA).
//   z       [B=16, T=2048, D=1024] f32
//   codebook[V=640, D=1024]        f32
//   W_p     [V=640, D=1024]        f32
//   gumbel  [B, T, V]              f32
// out: quantized [B,T,D] f32  ++  diversity_loss (1 f32)
//
// forward == { idx = argmax(z@cb^T + gumbel);  quantized = W_p[idx];
//              entropy of per-t histogram of idx over batch }
//
// Pipeline:
//   k0: codebook f32 -> bf16, pre-swizzled into WMMA B-fragment order (ws)
//   k1: bf16-WMMA logits GEMM fused with +gumbel and per-row argmax -> idx (ws)
//   k2: quantized[row,:] = W_p[idx[row],:]   (gather, W_p L2-hot)
//   k3: per-t histogram entropy -> scalar
// ---------------------------------------------------------------------------

typedef __attribute__((ext_vector_type(16))) __bf16 v16bf;
typedef __attribute__((ext_vector_type(8)))  float  v8f;

#define B_  16
#define T_  2048
#define D_  1024
#define V_  640
#define BT_ (B_ * T_)
#define EPS_ 1e-10f

#define NTILES_ (V_ / 16)   // 40 N-tiles of 16 columns
#define KSTEPS_ (D_ / 32)   // 32 K-steps of 32

__device__ __forceinline__ __bf16 f2bf(float x) { return (__bf16)x; }

// Per-lane K mapping for 16-bit WMMA A/B fragments (ISA 7.12.2, 16x32):
//   frag[j] holds K = (j/8)*16 + half*8 + (j%8), lane%16 = M (A) / N (B).

// Load one A fragment from f32 memory (base already includes +half*8) and
// convert to bf16 in registers.
__device__ __forceinline__ v16bf load_frag_bf16(const float* __restrict__ base) {
  float4 x0 = *(const float4*)(base +  0);
  float4 x1 = *(const float4*)(base +  4);
  float4 x2 = *(const float4*)(base + 16);
  float4 x3 = *(const float4*)(base + 20);
  v16bf f;
  f[0]  = f2bf(x0.x); f[1]  = f2bf(x0.y); f[2]  = f2bf(x0.z); f[3]  = f2bf(x0.w);
  f[4]  = f2bf(x1.x); f[5]  = f2bf(x1.y); f[6]  = f2bf(x1.z); f[7]  = f2bf(x1.w);
  f[8]  = f2bf(x2.x); f[9]  = f2bf(x2.y); f[10] = f2bf(x2.z); f[11] = f2bf(x2.w);
  f[12] = f2bf(x3.x); f[13] = f2bf(x3.y); f[14] = f2bf(x3.z); f[15] = f2bf(x3.w);
  return f;
}

// ---------------------------------------------------------------------------
// Kernel 0: codebook f32 -> bf16, stored in B-fragment order:
//   cbw[ ((nt*KSTEPS_ + k0s)*32 + lane) * 16 + j ]
// so the GEMM reads one contiguous 32B fragment per lane (2x b128, coalesced).
// One thread per fragment slot (40*32*32 = 40960 threads).
// ---------------------------------------------------------------------------
__global__ __launch_bounds__(256) void vq_cb_swizzle(
    const float* __restrict__ cb, __bf16* __restrict__ cbw)
{
  const int t = blockIdx.x * 256 + threadIdx.x;
  if (t >= NTILES_ * KSTEPS_ * 32) return;
  const int lane = t & 31;
  const int k0s  = (t >> 5) & (KSTEPS_ - 1);
  const int nt   = t >> 10;
  const int ln   = lane & 15;
  const int half = lane >> 4;
  const float* __restrict__ src =
      cb + (size_t)(nt * 16 + ln) * D_ + k0s * 32 + half * 8;
  __bf16* __restrict__ dst = cbw + (size_t)t * 16;
  #pragma unroll
  for (int j = 0; j < 16; ++j) {
    const int off = (j >> 3) * 16 + (j & 7);  // K(j) - half*8
    dst[j] = f2bf(src[off]);
  }
}

// ---------------------------------------------------------------------------
// Kernel 1: logits GEMM (bf16 WMMA) fused with +gumbel and row argmax.
// Grid: BT/32 workgroups of 256 threads (8 wave32s).
//   WG  -> 32 rows of z x all 640 codebook columns
//   wave-> 80-column slab (5 N-tiles of 16), 2 M-tiles of 16
// ---------------------------------------------------------------------------
__global__ __launch_bounds__(256) void vq_logits_argmax(
    const float* __restrict__ z, const __bf16* __restrict__ cbw,
    const float* __restrict__ gumbel, int* __restrict__ idx_out)
{
  const int lane = threadIdx.x & 31;
  const int wave = threadIdx.x >> 5;
  const int half = lane >> 4;     // which 16-lane half of the wave
  const int ln   = lane & 15;     // M index (A frag) / N index (C/D)
  const int rowbase = blockIdx.x * 32;
  const int nbase   = wave * 80;

  const v16bf* __restrict__ cbw_f = (const v16bf*)cbw;

  v8f acc[2][5];
  #pragma unroll
  for (int mt = 0; mt < 2; ++mt)
    #pragma unroll
    for (int nt = 0; nt < 5; ++nt)
      #pragma unroll
      for (int e = 0; e < 8; ++e) acc[mt][nt][e] = 0.0f;

  for (int k0s = 0; k0s < KSTEPS_; ++k0s) {
    const int k0 = k0s * 32;
    // A: f32 loads + in-register bf16 conversion (z is read exactly once).
    v16bf afrag[2];
    #pragma unroll
    for (int mt = 0; mt < 2; ++mt) {
      const float* base =
          z + (size_t)(rowbase + mt * 16 + ln) * D_ + k0 + half * 8;
      afrag[mt] = load_frag_bf16(base);
      __builtin_prefetch(base + 32, 0, 0);   // next K slab -> global_prefetch_b8
    }
    // B: pre-swizzled bf16 fragments, one 32B coalesced load per lane.
    v16bf bfrag[5];
    #pragma unroll
    for (int nt = 0; nt < 5; ++nt)
      bfrag[nt] =
          cbw_f[((size_t)(wave * 5 + nt) * KSTEPS_ + k0s) * 32 + lane];

    #pragma unroll
    for (int mt = 0; mt < 2; ++mt)
      #pragma unroll
      for (int nt = 0; nt < 5; ++nt)
        acc[mt][nt] = __builtin_amdgcn_wmma_f32_16x16x32_bf16(
            /*neg_a=*/false, afrag[mt], /*neg_b=*/false, bfrag[nt],
            /*c_mod=*/(short)0, acc[mt][nt],
            /*reuse_a=*/false, /*reuse_b=*/false);
  }

  // Epilogue: add gumbel (tau==1), per-lane partial argmax over this wave's
  // 80 columns, then cross-lane/cross-wave reduction through LDS.
  // C layout: acc[mt][nt][r] is row (mt*16 + r + 8*half), col (nbase+nt*16+ln).
  __shared__ float s_val[8][32][16];
  __shared__ int   s_idx[8][32][16];

  #pragma unroll
  for (int mt = 0; mt < 2; ++mt) {
    #pragma unroll
    for (int r = 0; r < 8; ++r) {
      const int row_local = mt * 16 + r + 8 * half;
      const size_t grow = (size_t)(rowbase + row_local) * V_;
      float bestv = -3.4e38f;
      int   besti = 0;
      #pragma unroll
      for (int nt = 0; nt < 5; ++nt) {
        const int v = nbase + nt * 16 + ln;
        const float s = acc[mt][nt][r] + gumbel[grow + v];
        if (s > bestv) { bestv = s; besti = v; }
      }
      s_val[wave][row_local][ln] = bestv;
      s_idx[wave][row_local][ln] = besti;
    }
  }
  __syncthreads();

  if (threadIdx.x < 32) {
    const int row_local = threadIdx.x;
    float bestv = -3.4e38f;
    int   besti = 0;
    #pragma unroll 2
    for (int w = 0; w < 8; ++w)
      for (int n = 0; n < 16; ++n) {
        const float v = s_val[w][row_local][n];
        if (v > bestv) { bestv = v; besti = s_idx[w][row_local][n]; }
      }
    idx_out[rowbase + row_local] = besti;
  }
}

// ---------------------------------------------------------------------------
// Kernel 2: quantized[row,:] = W_p[idx[row],:]  (pure gather; W_p is L2-hot)
// Grid: BT blocks x 256 threads; each thread moves one float4 (256*4 = 1024).
// ---------------------------------------------------------------------------
__global__ __launch_bounds__(256) void vq_gather(
    const int* __restrict__ idx, const float* __restrict__ Wp,
    float* __restrict__ out)
{
  const int row = blockIdx.x;
  const int v = idx[row];
  const float4* __restrict__ src = (const float4*)(Wp + (size_t)v * D_);
  float4* __restrict__ dst = (float4*)(out + (size_t)row * D_);
  dst[threadIdx.x] = src[threadIdx.x];
}

// ---------------------------------------------------------------------------
// Kernel 3: diversity loss. avg_probs[t,v] = count_b(idx[b,t]==v)/16;
// loss = mean_t( -sum_v p*log(p+eps) ). Only distinct present values matter.
// Single workgroup; writes the final scalar (also clears the 0xAA poison).
// ---------------------------------------------------------------------------
__global__ __launch_bounds__(256) void vq_entropy(
    const int* __restrict__ idx, float* __restrict__ out_scalar)
{
  __shared__ float red[256];
  float sum = 0.0f;
  for (int t = threadIdx.x; t < T_; t += 256) {
    int ids[B_];
    #pragma unroll
    for (int b = 0; b < B_; ++b) ids[b] = idx[b * T_ + t];
    float ent = 0.0f;
    #pragma unroll
    for (int b = 0; b < B_; ++b) {
      int c = 0;
      bool first = true;
      #pragma unroll
      for (int b2 = 0; b2 < B_; ++b2) {
        if (ids[b2] == ids[b]) { c++; if (b2 < b) first = false; }
      }
      if (first) {
        const float p = (float)c * (1.0f / (float)B_);
        ent -= p * __logf(p + EPS_);
      }
    }
    sum += ent;
  }
  red[threadIdx.x] = sum;
  __syncthreads();
  for (int s = 128; s > 0; s >>= 1) {
    if (threadIdx.x < (unsigned)s) red[threadIdx.x] += red[threadIdx.x + s];
    __syncthreads();
  }
  if (threadIdx.x == 0) out_scalar[0] = red[0] * (1.0f / (float)T_);
}

// ---------------------------------------------------------------------------
extern "C" void kernel_launch(void* const* d_in, const int* in_sizes, int n_in,
                              void* d_out, int out_size, void* d_ws, size_t ws_size,
                              hipStream_t stream) {
  (void)in_sizes; (void)n_in; (void)out_size; (void)ws_size;
  const float* z   = (const float*)d_in[0];
  const float* cb  = (const float*)d_in[1];
  const float* Wp  = (const float*)d_in[2];
  const float* gum = (const float*)d_in[3];
  float* out = (float*)d_out;

  // Workspace layout: [0, 128KB) idx int[BT]; [128KB, ...) bf16 codebook frags.
  int*    idx = (int*)d_ws;
  __bf16* cbw = (__bf16*)((char*)d_ws + (size_t)BT_ * sizeof(int));

  vq_cb_swizzle<<<(NTILES_ * KSTEPS_ * 32 + 255) / 256, 256, 0, stream>>>(cb, cbw);
  vq_logits_argmax<<<BT_ / 32, 256, 0, stream>>>(z, cbw, gum, idx);
  vq_gather<<<BT_, 256, 0, stream>>>(idx, Wp, out);
  vq_entropy<<<1, 256, 0, stream>>>(idx, out + (size_t)BT_ * D_);
}
